// MultiWindowDINEncoder_24026047054163
// MI455X (gfx1250) — matrix operand order repaired and checked by reading, not compiled, not run on hardware
//
#include <hip/hip_runtime.h>
#include <hip/hip_bf16.h>

typedef __attribute__((ext_vector_type(16))) __bf16 v16bf;
typedef __attribute__((ext_vector_type(8)))  float  v8f;
typedef __attribute__((ext_vector_type(4)))  int    v4i;

#define LL 200
#define LP 208          // padded rows
#define DD 64

static __device__ __forceinline__ v8f wmma_bf16(v16bf a, v16bf b, v8f c) {
  // (neg_a, A, neg_b, B, c_mod, C, reuse_a, reuse_b)
  return __builtin_amdgcn_wmma_f32_16x16x32_bf16(false, a, false, b, (short)0, c, false, false);
}

// Build 16x32 bf16 A-fragment (hi/lo split) for this lane from a row of floats.
// Lane's 16 elements are two contiguous runs of 8: [c0..c0+7] and [c0+16..c0+23].
static __device__ __forceinline__ void build_a(const float* __restrict__ base, int c0,
                                               v16bf& ah, v16bf& al) {
  const float4 x0 = *(const float4*)(base + c0);
  const float4 x1 = *(const float4*)(base + c0 + 4);
  const float4 x2 = *(const float4*)(base + c0 + 16);
  const float4 x3 = *(const float4*)(base + c0 + 20);
  const float xs[16] = {x0.x, x0.y, x0.z, x0.w, x1.x, x1.y, x1.z, x1.w,
                        x2.x, x2.y, x2.z, x2.w, x3.x, x3.y, x3.z, x3.w};
  #pragma unroll
  for (int j = 0; j < 16; ++j) {
    __bf16 h = (__bf16)xs[j];
    ah[j] = h;
    al[j] = (__bf16)(xs[j] - (float)h);
  }
}

static __device__ __forceinline__ unsigned short bf_hi(float v) {
  union { __bf16 b; unsigned short u; } c; c.b = (__bf16)v; return c.u;
}
static __device__ __forceinline__ unsigned short bf_lo(float v) {
  union { __bf16 b; unsigned short u; } c; c.b = (__bf16)v;
  union { __bf16 b; unsigned short u; } l; l.b = (__bf16)(v - (float)c.b); return l.u;
}

// ---------------------------------------------------------------------------
// Pack W2 (64x32) into bf16 hi/lo WMMA B-fragments in ws.
// B-fragment (ISA 7.12.2, 16-bit 32x16 B): lane = N%16 + 16*((K/8)%2), elem j = K%8 + 8*(K/16).
// ws (ushort units): W2H @0 [2048], W2L @2048.
// ---------------------------------------------------------------------------
__global__ __launch_bounds__(256) void pack_weights_kernel(
    const float* __restrict__ W2, unsigned short* __restrict__ ws)
{
  int t = blockIdx.x * blockDim.x + threadIdx.x;   // 0 .. 2047
  int j = t & 15, lane = (t >> 4) & 31, nt = (t >> 9) & 1, kt = (t >> 10) & 1;
  int k = kt * 32 + (j & 7) + ((lane & 16) >> 1) + ((j & 8) << 1);
  int n = nt * 16 + (lane & 15);
  float v = W2[k * 32 + n];
  ws[t] = bf_hi(v);
  ws[2048 + t] = bf_lo(v);
}

// ---------------------------------------------------------------------------
// Main: one 128-thread (4-wave) workgroup per batch element.
// attn_input @ W1 = seq @ (W1[:64] + diag(q) @ W1[64:128]) + const(q) section,
// so layer 1 collapses to a single 64-K GEMM against per-block weights Wc.
// ---------------------------------------------------------------------------
__global__ __launch_bounds__(128, 4) void din_main_kernel(
    const float* __restrict__ query, const float* __restrict__ seq,
    const int* __restrict__ slenp, const float* __restrict__ W1,
    const float* __restrict__ b1, const float* __restrict__ b2,
    const float* __restrict__ Wl, const float* __restrict__ blp,
    const float* __restrict__ pap,
    const unsigned short* __restrict__ ws,
    float* __restrict__ out)
{
  __shared__ float sSeq[LP * DD];              // 53248 B (rows 200..207 zeroed)
  __shared__ float sQ[DD];                     //   256 B
  __shared__ float sQC[DD];                    //   256 B  qc[n] = sum_k q[k]*W1[128+k][n]
  __shared__ unsigned short sWCH[4096];        //  8192 B  Wc hi B-fragments
  __shared__ unsigned short sWCL[4096];        //  8192 B  Wc lo B-fragments
  __shared__ float sH1[4][16 * 64];            // 16384 B
  __shared__ float sH2[4][16 * 32];            //  8192 B
  __shared__ float sAttn[LP];                  //   832 B

  const int tid    = threadIdx.x;
  const int b      = blockIdx.x;
  const int lane   = tid & 31;
  const int wave   = tid >> 5;
  const int hof    = (lane & 16) >> 1;         // 8 for lanes 16..31 (A-layout K offset)
  const int ln     = lane & 15;
  const int slen   = slenp[b];

  // ---- stage sequence (+ zero pad) and query ----
  const float4* seqB = (const float4*)(seq + (size_t)b * (LL * DD));
#if defined(__gfx1250__) && __has_builtin(__builtin_amdgcn_global_load_async_to_lds_b128)
  for (int i = tid; i < (LL * DD / 4); i += 128)
    __builtin_amdgcn_global_load_async_to_lds_b128(
        (__attribute__((address_space(1))) v4i*)(seqB + i),
        (__attribute__((address_space(3))) v4i*)(sSeq + 4 * i),
        0, 0);
#else
  for (int i = tid; i < (LL * DD / 4); i += 128) ((float4*)sSeq)[i] = seqB[i];
#endif
  ((float4*)sSeq)[LL * DD / 4 + tid] = make_float4(0.f, 0.f, 0.f, 0.f); // 128 vecs = 8 rows
  if (tid < DD) sQ[tid] = query[b * DD + tid];
#if defined(__gfx1250__) && __has_builtin(__builtin_amdgcn_global_load_async_to_lds_b128)
#if __has_builtin(__builtin_amdgcn_s_wait_asynccnt)
  __builtin_amdgcn_s_wait_asynccnt(0);
#else
  asm volatile("s_wait_asynccnt 0" ::: "memory");
#endif
#endif
  __syncthreads();

  // ---- per-block: Wc[k][n] = W1[k][n] + q[k]*W1[64+k][n] -> bf16 B-frags;
  //      qc[n] = sum_k q[k]*W1[128+k][n].  Loop over (k,n), n fastest, so the
  //      W1 reads are coalesced; scatter into fragment order in LDS. ----
  for (int idx = tid; idx < 4096; idx += 128) {
    int k = idx >> 6;                          // 0..63
    int n = idx & 63;
    float v = W1[k * 64 + n] + sQ[k] * W1[(64 + k) * 64 + n];
    int e = (((k >> 5) * 4 + (n >> 4)) * 32 + (n & 15) + ((k & 8) << 1)) * 16 + ((k & 7) | ((k & 16) >> 1));
    sWCH[e] = bf_hi(v);
    sWCL[e] = bf_lo(v);
  }
  if (tid < DD) {
    float a = 0.f;
    for (int k = 0; k < DD; ++k) a += sQ[k] * W1[(128 + k) * 64 + tid];
    sQC[tid] = a;
  }
  __syncthreads();

  // ---- attention MLP over 16-row tiles (13 tiles, wave-strided) ----
  for (int t = wave; t < 13; t += 4) {
    const int row = t * 16 + ln;               // < 208, padded rows are zero

    // ----- layer 1: (16x64 seq) @ (64x64 Wc) + (b1 + qc), ReLU -----
    v8f acc[4];
    #pragma unroll
    for (int nt = 0; nt < 4; ++nt) {
      float bn = b1[nt * 16 + ln] + sQC[nt * 16 + ln];
      #pragma unroll
      for (int r = 0; r < 8; ++r) acc[nt][r] = bn;
    }
    #pragma unroll
    for (int kt = 0; kt < 2; ++kt) {
      v16bf ah, al;
      build_a(&sSeq[row * 64], kt * 32 + hof, ah, al);
      #pragma unroll
      for (int nt = 0; nt < 4; ++nt) {
        const int fo = ((kt * 4 + nt) * 32 + lane) * 16;
        const v16bf bh = *(const v16bf*)(sWCH + fo);
        const v16bf bl = *(const v16bf*)(sWCL + fo);
        acc[nt] = wmma_bf16(ah, bh, acc[nt]);
        acc[nt] = wmma_bf16(ah, bl, acc[nt]);
        acc[nt] = wmma_bf16(al, bh, acc[nt]);
      }
    }
    #pragma unroll
    for (int nt = 0; nt < 4; ++nt)
      #pragma unroll
      for (int r = 0; r < 8; ++r) {
        int m = r + ((lane & 16) >> 1);        // C/D layout: lanes16-31 hold M+8
        float v = acc[nt][r];
        sH1[wave][m * 64 + nt * 16 + ln] = v > 0.f ? v : 0.f;
      }

    // ----- layer 2: (16x64) @ (64x32) + b2, ReLU -----
    v8f acc2[2];
    #pragma unroll
    for (int nt = 0; nt < 2; ++nt) {
      float bn = b2[nt * 16 + ln];
      #pragma unroll
      for (int r = 0; r < 8; ++r) acc2[nt][r] = bn;
    }
    #pragma unroll
    for (int kt = 0; kt < 2; ++kt) {
      v16bf ah, al;
      build_a(&sH1[wave][ln * 64], kt * 32 + hof, ah, al);
      #pragma unroll
      for (int nt = 0; nt < 2; ++nt) {
        const int fo = ((kt * 2 + nt) * 32 + lane) * 16;
        const v16bf bh = *(const v16bf*)(ws + fo);
        const v16bf bl = *(const v16bf*)(ws + 2048 + fo);
        acc2[nt] = wmma_bf16(ah, bh, acc2[nt]);
        acc2[nt] = wmma_bf16(ah, bl, acc2[nt]);
        acc2[nt] = wmma_bf16(al, bh, acc2[nt]);
      }
    }
    #pragma unroll
    for (int nt = 0; nt < 2; ++nt)
      #pragma unroll
      for (int r = 0; r < 8; ++r) {
        int m = r + ((lane & 16) >> 1);
        float v = acc2[nt][r];
        sH2[wave][m * 32 + nt * 16 + ln] = v > 0.f ? v : 0.f;
      }

    // ----- layer 3: (16x32) @ (32x1) + bl, PReLU, mask -----
    if (lane < 16) {
      int l = t * 16 + lane;
      float a = blp[0];
      #pragma unroll
      for (int k = 0; k < 32; ++k) a += sH2[wave][lane * 32 + k] * Wl[k];
      a = a >= 0.f ? a : pap[0] * a;
      if (l >= slen) a = 0.f;                  // mask (also zeroes pad rows)
      sAttn[l] = a;
    }
  }
  __syncthreads();

  // ---- windowed mean-pooling + query copy ----
  const int cumw[5] = {0, 10, 30, 60, 110};
  const int winw[5] = {10, 20, 30, 50, 90};
  float* outB = out + (size_t)b * 384;
  for (int o = tid; o < 320; o += 128) {
    int w = o >> 6, d = o & 63;
    float s = 0.f;
    int st = cumw[w], en = st + winw[w];
    for (int l = st; l < en; ++l) s += sAttn[l] * sSeq[l * 64 + d];
    int seg = slen - st;
    if (seg > winw[w]) seg = winw[w];
    if (seg < 1) seg = 1;
    outB[o] = s / (float)seg;
  }
  if (tid < 64) outB[320 + tid] = sQ[tid];
}

extern "C" void kernel_launch(void* const* d_in, const int* in_sizes, int n_in,
                              void* d_out, int out_size, void* d_ws, size_t ws_size,
                              hipStream_t stream) {
  (void)in_sizes; (void)n_in; (void)out_size; (void)ws_size;
  const float* query = (const float*)d_in[0];
  const float* seq   = (const float*)d_in[1];
  const int*   slen  = (const int*)d_in[2];
  const float* W1    = (const float*)d_in[3];
  const float* b1    = (const float*)d_in[4];
  const float* W2    = (const float*)d_in[5];
  const float* b2    = (const float*)d_in[6];
  const float* Wl    = (const float*)d_in[7];
  const float* bl    = (const float*)d_in[8];
  const float* pa    = (const float*)d_in[9];
  unsigned short* wsu = (unsigned short*)d_ws;   // needs 8192 bytes

  pack_weights_kernel<<<8, 256, 0, stream>>>(W2, wsu);
  din_main_kernel<<<4096, 128, 0, stream>>>(query, seq, slen, W1, b1, b2, Wl, bl, pa,
                                            wsu, (float*)d_out);
}
